// Atts_11751030522296
// MI455X (gfx1250) — matrix operand order repaired and checked by reading, not compiled, not run on hardware
//
#include <hip/hip_runtime.h>
#include <math.h>

typedef __attribute__((ext_vector_type(16))) _Float16 v16h;
typedef __attribute__((ext_vector_type(8)))  _Float16 h8;
typedef __attribute__((ext_vector_type(2)))  _Float16 h2;
typedef __attribute__((ext_vector_type(8)))  float    v8f;
typedef __attribute__((ext_vector_type(4)))  float    f4;
typedef __attribute__((ext_vector_type(2)))  float    f2;

// ---------------------------------------------------------------------------
// LayerNorm over channel dim, one wave per token row.
// nchw==1: x is [B,C,Npos], token (b=blockIdx.y, n); grid=(ceil(Npos/8), B).
// nchw==0: x is row-major [R,C]; grid=(ceil(R/8), 1), Npos=R.
// Output row-major [R,C] (row = b*Npos + n). All index math fits in int32.
// ---------------------------------------------------------------------------
__global__ void ln_kernel(const float* __restrict__ x, const float* __restrict__ g,
                          const float* __restrict__ bta, float* __restrict__ y,
                          int C, int Npos, int nchw) {
  int wave = threadIdx.x >> 5;
  int lane = threadIdx.x & 31;
  int nb   = blockIdx.x * (blockDim.x >> 5) + wave;
  if (nb >= Npos) return;
  int b = blockIdx.y;
  int row  = b * Npos + nb;
  int base = nchw ? (b * C * Npos + nb) : (row * C);
  int cstr = nchw ? Npos : 1;
  float s = 0.f, ss = 0.f;
  for (int c = lane; c < C; c += 32) {
    float v = x[base + c * cstr];
    s += v; ss += v * v;
  }
#pragma unroll
  for (int m = 1; m <= 16; m <<= 1) {
    s  += __shfl_xor(s,  m, 32);
    ss += __shfl_xor(ss, m, 32);
  }
  float mu   = s / (float)C;
  float var  = ss / (float)C - mu * mu;
  float rstd = rsqrtf(var + 1e-5f);
  for (int c = lane; c < C; c += 32) {
    float v = x[base + c * cstr];
    y[row * C + c] = (v - mu) * rstd * g[c] + bta[c];
  }
}

// ---------------------------------------------------------------------------
// Sparse select (templated on kernel size KK so all divisions are by
// constants; C is a power of two -> shift/mask via cshift).
// One block per (b, patch); Hk = Wk = 12 for every scale, L = 144.
// ---------------------------------------------------------------------------
template <int KK>
__global__ void select_kernel(const float* __restrict__ x, float* __restrict__ sel,
                              int C, int cshift, int H, int W, int topk) {
  constexpr int V = KK * KK;
  __shared__ float u[1664];
  __shared__ float s[256];
  __shared__ float rel[64];
  __shared__ int   idx[8];
  const int L = 144;
  int bl = blockIdx.x;
  int b  = bl / L, l = bl - b * L;
  int hk = l / 12, wk = l - hk * 12;
  int tid = threadIdx.x;
  int xb = b * C * H * W;
  for (int t = tid; t < V * C; t += blockDim.x) {
    int vv = t >> cshift, c = t & (C - 1);
    int kh = vv / KK, kw = vv - kh * KK;
    u[t] = x[xb + c * H * W + (hk * KK + kh) * W + (wk * KK + kw)];
  }
  __syncthreads();
  for (int c = tid; c < C; c += blockDim.x) {
    float acc = 0.f;
#pragma unroll 1
    for (int vv = 0; vv < V; ++vv) acc += u[vv * C + c];
    s[c] = acc;
  }
  __syncthreads();
  for (int vv = tid; vv < V; vv += blockDim.x) {
    float acc = 0.f;
    for (int c = 0; c < C; ++c) acc += u[vv * C + c] * s[c];
    rel[vv] = acc / (float)V;
  }
  __syncthreads();
  if (tid == 0) {
    unsigned long long used = 0ull;
    for (int t = 0; t < topk; ++t) {
      float best = 3.4e38f; int bi = 0;
      for (int vv = 0; vv < V; ++vv) {
        if ((used >> vv) & 1ull) continue;
        if (rel[vv] < best) { best = rel[vv]; bi = vv; }
      }
      used |= (1ull << bi);
      idx[t] = bi;
    }
  }
  __syncthreads();
  for (int t = tid; t < topk * C; t += blockDim.x) {
    int tt = t >> cshift, c = t & (C - 1);
    sel[((b * L + l) * topk + tt) * C + c] = u[idx[tt] * C + c];
  }
}

// ---------------------------------------------------------------------------
// WMMA GEMM: Y = X(f32,[R,K] rm) * W(f32,[K,Cout] rm), Y f32.
// R % 16 == 0, K % 32 == 0, Cout % 16 == 0.
// Block: 256 threads = 8 waves; each wave owns a 16-col slice of a 16x128 tile.
// B tile staged TRANSPOSED in LDS ([col][k]) so each lane's WMMA fragment is
// two ds_load_b128's.
// Normal mode:    grid=(R/16, colTiles, 1), Ntok=0.
// Transpose mode: grid=(Ntok/16, colTiles, B); Y[b, col, ntok]. No divisions.
// ---------------------------------------------------------------------------
__global__ void gemm_wmma_kernel(const float* __restrict__ X, const float* __restrict__ Wm,
                                 float* __restrict__ Y, int K, int Cout,
                                 int transpose_out, int Ntok) {
  __shared__ _Float16 As[16 * 32];    // [row][k]
  __shared__ _Float16 Bst[128 * 32];  // [col][k]  (transposed tile)
  int tid  = threadIdx.x;
  int wave = tid >> 5, lane = tid & 31;
  int hseg = lane >> 4, n = lane & 15;
  int rbase = blockIdx.z * Ntok + blockIdx.x * 16;
  int cbase = blockIdx.y * 128;
  int wcol  = cbase + wave * 16;
  bool active = (wcol < Cout);
  v8f acc = {};
  for (int kc = 0; kc < K; kc += 32) {
    {  // A tile: float2 loads, packed half2 stores
      int r  = tid >> 4;
      int c2 = (tid & 15) * 2;
      f2 xv = *(const f2*)(&X[(rbase + r) * K + kc + c2]);
      h2 hv; hv[0] = (_Float16)xv[0]; hv[1] = (_Float16)xv[1];
      *(h2*)(&As[r * 32 + c2]) = hv;
    }
#pragma unroll
    for (int sld = 0; sld < 16; ++sld) {  // B tile, transposed store
      int e = tid + 256 * sld;
      int r = e >> 7, c = e & 127;
      int col = cbase + c;
      Bst[c * 32 + r] = (col < Cout) ? (_Float16)Wm[(kc + r) * Cout + col]
                                     : (_Float16)0.f;
    }
    if (kc + 32 < K) {  // prefetch next W k-chunk (global_prefetch_b8)
      __builtin_prefetch(&Wm[(kc + 32 + (tid & 31)) * Cout + cbase], 0, 1);
    }
    __syncthreads();
    if (active) {
      h8 alo = *(const h8*)(&As[n * 32 + 8 * hseg]);
      h8 ahi = *(const h8*)(&As[n * 32 + 16 + 8 * hseg]);
      v16h a;
#pragma unroll
      for (int e = 0; e < 8; ++e) { a[e] = alo[e]; a[8 + e] = ahi[e]; }
      h8 blo = *(const h8*)(&Bst[(wave * 16 + n) * 32 + 8 * hseg]);
      h8 bhi = *(const h8*)(&Bst[(wave * 16 + n) * 32 + 16 + 8 * hseg]);
      v16h bf;
#pragma unroll
      for (int e = 0; e < 8; ++e) { bf[e] = blo[e]; bf[8 + e] = bhi[e]; }
      acc = __builtin_amdgcn_wmma_f32_16x16x32_f16(false, a, false, bf, (short)0, acc,
                                                   false, false);
    }
    __syncthreads();
  }
  if (active) {
    if (!transpose_out) {
#pragma unroll
      for (int r = 0; r < 8; ++r) {
        int row = rbase + hseg * 8 + r;
        Y[row * Cout + wcol + n] = acc[r];
      }
    } else {
      int bq  = blockIdx.z;
      int ntb = blockIdx.x * 16;
      float* yb = Y + bq * Cout * Ntok + (wcol + n) * Ntok + ntb;
#pragma unroll
      for (int r = 0; r < 8; ++r) yb[hseg * 8 + r] = acc[r];
    }
  }
}

// ---------------------------------------------------------------------------
// Flash attention for one (query scale, kv source): 4 waves/block = 4 heads,
// block handles 16 query rows. Online softmax, all matmuls via WMMA f16->f32.
// qp:  [B*Nq, 4D]  (head h at cols h*D..)
// kvp: [B*M, 8D]   (k at h*D, v at 4D + h*D)
// outc:[B*Nq, 12D] write slice at h*3D + src*D
// ---------------------------------------------------------------------------
template <int D>
__global__ void attn_kernel(const float* __restrict__ qp, const float* __restrict__ kvp,
                            float* __restrict__ outc, int Nq, int M, int src, float scale) {
  constexpr int KC = D / 32;  // k-chunks over head dim
  constexpr int OG = D / 16;  // output 16-col groups
  __shared__ _Float16 Plds[4][16 * 32];
  int tid  = threadIdx.x;
  int h    = tid >> 5;
  int lane = tid & 31;
  int hseg = lane >> 4, n = lane & 15;
  int qbase = blockIdx.x * 16;
  int b     = blockIdx.y;
  const int ldq = 4 * D, ldkv = 8 * D, ldo = 12 * D;

  v16h qf[KC];
#pragma unroll
  for (int kc = 0; kc < KC; ++kc) {
    const float* qr = qp + (b * Nq + qbase + n) * ldq + h * D + kc * 32 + 16 * hseg;
    f4 q0 = *(const f4*)(qr);
    f4 q1 = *(const f4*)(qr + 4);
    f4 q2 = *(const f4*)(qr + 8);
    f4 q3 = *(const f4*)(qr + 12);
#pragma unroll
    for (int e = 0; e < 4; ++e) {
      qf[kc][e]      = (_Float16)q0[e];
      qf[kc][4 + e]  = (_Float16)q1[e];
      qf[kc][8 + e]  = (_Float16)q2[e];
      qf[kc][12 + e] = (_Float16)q3[e];
    }
  }

  v8f o[OG];
#pragma unroll
  for (int g = 0; g < OG; ++g) o[g] = (v8f){};
  float rmax[8], rsum[8];
#pragma unroll
  for (int r = 0; r < 8; ++r) { rmax[r] = -1e38f; rsum[r] = 0.f; }

  int mtiles = (M + 31) >> 5;
  for (int mt = 0; mt < mtiles; ++mt) {
    int mbase = mt * 32;
    int key0 = mbase + n, key1 = mbase + 16 + n;
    bool ok0 = key0 < M, ok1 = key1 < M;
    v8f s0 = {}, s1 = {};
#pragma unroll
    for (int kc = 0; kc < KC; ++kc) {
      v16h bk0, bk1;
      if (ok0) {
        const float* k0 = kvp + (b * M + key0) * ldkv + h * D + kc * 32 + 16 * hseg;
        f4 a0 = *(const f4*)(k0), a1 = *(const f4*)(k0 + 4);
        f4 a2 = *(const f4*)(k0 + 8), a3 = *(const f4*)(k0 + 12);
#pragma unroll
        for (int e = 0; e < 4; ++e) {
          bk0[e] = (_Float16)a0[e]; bk0[4 + e] = (_Float16)a1[e];
          bk0[8 + e] = (_Float16)a2[e]; bk0[12 + e] = (_Float16)a3[e];
        }
      } else {
#pragma unroll
        for (int e = 0; e < 16; ++e) bk0[e] = (_Float16)0.f;
      }
      if (ok1) {
        const float* k1 = kvp + (b * M + key1) * ldkv + h * D + kc * 32 + 16 * hseg;
        f4 a0 = *(const f4*)(k1), a1 = *(const f4*)(k1 + 4);
        f4 a2 = *(const f4*)(k1 + 8), a3 = *(const f4*)(k1 + 12);
#pragma unroll
        for (int e = 0; e < 4; ++e) {
          bk1[e] = (_Float16)a0[e]; bk1[4 + e] = (_Float16)a1[e];
          bk1[8 + e] = (_Float16)a2[e]; bk1[12 + e] = (_Float16)a3[e];
        }
      } else {
#pragma unroll
        for (int e = 0; e < 16; ++e) bk1[e] = (_Float16)0.f;
      }
      s0 = __builtin_amdgcn_wmma_f32_16x16x32_f16(false, qf[kc], false, bk0, (short)0, s0,
                                                  false, false);
      s1 = __builtin_amdgcn_wmma_f32_16x16x32_f16(false, qf[kc], false, bk1, (short)0, s1,
                                                  false, false);
    }
    float p0[8], p1[8];
#pragma unroll
    for (int r = 0; r < 8; ++r) {
      float a0 = ok0 ? s0[r] * scale : -1e30f;
      float a1 = ok1 ? s1[r] * scale : -1e30f;
      float t = fmaxf(a0, a1);
#pragma unroll
      for (int msk = 1; msk <= 8; msk <<= 1) t = fmaxf(t, __shfl_xor(t, msk, 32));
      float nm    = fmaxf(rmax[r], t);
      float alpha = __expf(rmax[r] - nm);
      rmax[r] = nm;
      float e0 = __expf(a0 - nm);
      float e1 = __expf(a1 - nm);
      p0[r] = e0; p1[r] = e1;
      float rs = e0 + e1;
#pragma unroll
      for (int msk = 1; msk <= 8; msk <<= 1) rs += __shfl_xor(rs, msk, 32);
      rsum[r] = rsum[r] * alpha + rs;
#pragma unroll
      for (int g = 0; g < OG; ++g) o[g][r] *= alpha;
    }
    __syncthreads();
#pragma unroll
    for (int r = 0; r < 8; ++r) {
      int row = hseg * 8 + r;
      Plds[h][row * 32 + n]      = (_Float16)p0[r];
      Plds[h][row * 32 + 16 + n] = (_Float16)p1[r];
    }
    __syncthreads();
    h8 plo = *(const h8*)(&Plds[h][n * 32 + 8 * hseg]);
    h8 phi = *(const h8*)(&Plds[h][n * 32 + 16 + 8 * hseg]);
    v16h ap;
#pragma unroll
    for (int e = 0; e < 8; ++e) { ap[e] = plo[e]; ap[8 + e] = phi[e]; }
#pragma unroll
    for (int g = 0; g < OG; ++g) {
      v16h bv;
#pragma unroll
      for (int e = 0; e < 16; ++e) {
        int key = mbase + (e < 8 ? 8 * hseg + e : 16 + 8 * hseg + (e - 8));
        bv[e] = (key < M)
                    ? (_Float16)kvp[(b * M + key) * ldkv + 4 * D + h * D + g * 16 + n]
                    : (_Float16)0.f;
      }
      o[g] = __builtin_amdgcn_wmma_f32_16x16x32_f16(false, ap, false, bv, (short)0, o[g],
                                                    false, false);
    }
  }
#pragma unroll
  for (int g = 0; g < OG; ++g) {
#pragma unroll
    for (int r = 0; r < 8; ++r) {
      int row = qbase + hseg * 8 + r;
      float val = o[g][r] / rsum[r];
      outc[(b * Nq + row) * ldo + h * 3 * D + src * D + g * 16 + n] = val;
    }
  }
}

// ---------------------------------------------------------------------------
extern "C" void kernel_launch(void* const* d_in, const int* in_sizes, int n_in,
                              void* d_out, int out_size, void* d_ws, size_t ws_size,
                              hipStream_t stream) {
  (void)in_sizes; (void)n_in; (void)out_size; (void)ws_size;
  const int B = 2, L = 144;
  const int Cs[4]  = {32, 64, 128, 256};
  const int CSh[4] = {5, 6, 7, 8};  // log2(C)
  const int Hs[4]  = {84, 60, 36, 12};
  const int TKs[4] = {4, 3, 3, 1};
  int Ns[4], selLen[4];
  for (int i = 0; i < 4; ++i) { Ns[i] = Hs[i] * Hs[i]; selLen[i] = L * TKs[i]; }

  float* ws = (float*)d_ws;
  size_t off = 0;
  size_t off_skip[4], off_sel[4];
  for (int i = 0; i < 4; ++i) { off_skip[i] = off; off += (size_t)B * Cs[i] * Ns[i]; }
  for (int i = 0; i < 4; ++i) { off_sel[i] = off; off += (size_t)B * selLen[i] * Cs[i]; }
  size_t off_qn  = off; off += 460800;   // max B*N*C
  size_t off_qp  = off; off += 1843200;  // max B*N*4C
  size_t off_kvn = off; off += 110592;   // max B*M*C_j
  size_t off_kvp = off; off += 2359296;  // max B*M*8C
  size_t off_cat = off; off += 5529600;  // max B*N*12C

  // params: leaves start at d_in[4]; per (layer,scale): 13 leaves in jax
  // sorted-key order: [Wout, Wq, kvW0,kvb0,kvg0, kvW1,kvb1,kvg1, kvW2,kvb2,kvg2, qb, qg]
  auto leaf = [&](int l, int i, int o) -> const float* {
    return (const float*)d_in[4 + (l * 4 + i) * 13 + o];
  };

  for (int l = 0; l < 2; ++l) {
    const float* xin[4];
    float* xout[4];
    for (int i = 0; i < 4; ++i) {
      xin[i]  = (l == 0) ? (const float*)d_in[i] : ws + off_skip[i];
      xout[i] = (l == 0) ? ws + off_skip[i] : ((float*)d_out) + off_skip[i];
    }
    // sparse select for all scales (from current skips)
    select_kernel<7><<<B * L, 128, 0, stream>>>(xin[0], ws + off_sel[0], Cs[0], CSh[0],
                                                Hs[0], Hs[0], TKs[0]);
    select_kernel<5><<<B * L, 128, 0, stream>>>(xin[1], ws + off_sel[1], Cs[1], CSh[1],
                                                Hs[1], Hs[1], TKs[1]);
    select_kernel<3><<<B * L, 128, 0, stream>>>(xin[2], ws + off_sel[2], Cs[2], CSh[2],
                                                Hs[2], Hs[2], TKs[2]);
    select_kernel<1><<<B * L, 128, 0, stream>>>(xin[3], ws + off_sel[3], Cs[3], CSh[3],
                                                Hs[3], Hs[3], TKs[3]);
    for (int i = 0; i < 4; ++i) {
      int C = Cs[i], N = Ns[i], R = B * N;
      // q path: LN (NCHW gather) + Wq projection
      {
        dim3 gl((N + 7) / 8, B);
        ln_kernel<<<gl, 256, 0, stream>>>(xin[i], leaf(l, i, 12), leaf(l, i, 11),
                                          ws + off_qn, C, N, 1);
      }
      {
        dim3 gq(R / 16, (4 * C + 127) / 128, 1);
        gemm_wmma_kernel<<<gq, 256, 0, stream>>>(ws + off_qn, leaf(l, i, 1), ws + off_qp, C,
                                                 4 * C, 0, 0);
      }
      float scale = 1.0f / sqrtf((float)C);
      int t = 0;
      for (int j = 0; j < 4; ++j) {
        if (j == i) continue;
        int Cj = Cs[j], Mj = selLen[j], Rk = B * Mj;
        {
          dim3 gl((Rk + 7) / 8, 1);
          ln_kernel<<<gl, 256, 0, stream>>>(ws + off_sel[j], leaf(l, i, 2 + 3 * t + 2),
                                            leaf(l, i, 2 + 3 * t + 1), ws + off_kvn, Cj, Rk,
                                            0);
        }
        {
          dim3 gk(Rk / 16, (8 * C + 127) / 128, 1);
          gemm_wmma_kernel<<<gk, 256, 0, stream>>>(ws + off_kvn, leaf(l, i, 2 + 3 * t),
                                                   ws + off_kvp, Cj, 8 * C, 0, 0);
        }
        dim3 ga(N / 16, B);
        switch (C) {
          case 32:
            attn_kernel<32><<<ga, 128, 0, stream>>>(ws + off_qp, ws + off_kvp, ws + off_cat,
                                                    N, Mj, t, scale);
            break;
          case 64:
            attn_kernel<64><<<ga, 128, 0, stream>>>(ws + off_qp, ws + off_kvp, ws + off_cat,
                                                    N, Mj, t, scale);
            break;
          case 128:
            attn_kernel<128><<<ga, 128, 0, stream>>>(ws + off_qp, ws + off_kvp, ws + off_cat,
                                                     N, Mj, t, scale);
            break;
          default:
            attn_kernel<256><<<ga, 128, 0, stream>>>(ws + off_qp, ws + off_kvp, ws + off_cat,
                                                     N, Mj, t, scale);
            break;
        }
        ++t;
      }
      // output projection with fused transpose back to [B, C, H, W]
      // grid.z = batch so the epilogue needs no division at all
      {
        dim3 go(N / 16, (C + 127) / 128, B);
        gemm_wmma_kernel<<<go, 256, 0, stream>>>(ws + off_cat, leaf(l, i, 0), xout[i], 12 * C,
                                                 C, 1, N);
      }
    }
  }
}
